// lstm_model_28879360098509
// MI455X (gfx1250) — compile-verified, matrix operand order checked by reading, not run on hardware
//
#include <hip/hip_runtime.h>
#include <cstddef>
#include <cstdint>

// ---------------------------------------------------------------------------
// Persona seq2seq LSTM for MI455X (gfx1250, wave32, WMMA).
//
// bf16 weights (decoder-step set ~99MB fits the 192MB global L2, so per-decoder-step
// weight re-reads hit L2 not HBM), v_wmma_f32_16x16x32_bf16 with fp32 accum,
// input-GEMM batching across time, decoder constant-part precompute
// (enc_flat/spk @ W once), W+U folding for decoder layers 2-4 (x==h there),
// LDS-tiled weight transpose, deterministic loss reduction.
//
// GEMM inner loop is software-pipelined (ping-pong fragment double buffer) so
// the compiler can issue partial s_wait_loadcnt instead of waiting to zero
// before each WMMA. Requires K % 64 == 0 (true for all call sites:
// K in {512, 1024, 10240}).
// ---------------------------------------------------------------------------

#define Bz 64
#define Ss 20
#define Tt 20
#define Ee 512
#define Hh 1024
#define Vv 32000
#define H4 4096          // 4*H

typedef __bf16 bf16;
typedef __attribute__((ext_vector_type(8)))  bf16  v8bf;
typedef __attribute__((ext_vector_type(16))) bf16  v16bf;
typedef __attribute__((ext_vector_type(8)))  float v8f;

union FragAB { v16bf v; v8bf h[2]; };

__device__ __forceinline__ float sigmoidf(float x) { return 1.0f / (1.0f + __expf(-x)); }

// ---------------------------------------------------------------------------
// WMMA GEMM: C(M,N) f32 = A(M,K)bf16 row-major @ B(K,N), with B supplied as
// BT(N,K) bf16 row-major. Optional: += Csrc, += bias[n], ReLU.
// One wave computes a 16x64 tile (4 N-subtiles, A fragment reused 4x).
// Fragment layout per CDNA5 ISA 7.12.2:
//   lane = 16*hi + r; A: row m0+r, elems 0..7 -> K k+8*hi.., 8..15 -> K k+16+8*hi..
//   BT row n0+r gives the B operand (column n fixed per lane, same K mapping).
//   C/D: elem v -> (row m0+v+8*hi, col n0+r).
// ---------------------------------------------------------------------------
__global__ __launch_bounds__(256)
void k_wmma_gemm(const bf16* __restrict__ A, const bf16* __restrict__ BT,
                 float* __restrict__ C, const float* __restrict__ Csrc,
                 const float* __restrict__ bias,
                 int M, int N, int K, int relu)
{
    const int NT = 4;
    int wid  = (blockIdx.x * 256 + threadIdx.x) >> 5;   // global wave id (wave32)
    int lane = threadIdx.x & 31;
    int tilesN = N / (16 * NT);
    int tilesM = M / 16;
    if (wid >= tilesM * tilesN) return;                  // whole-wave exit: EXEC stays all-ones
    int tm = wid / tilesN, tn = wid % tilesN;
    int m0 = tm * 16, n0 = tn * (16 * NT);
    int r  = lane & 15, hi = lane >> 4;

    v8f acc[NT];
    const v8f vz = {0.f, 0.f, 0.f, 0.f, 0.f, 0.f, 0.f, 0.f};
#pragma unroll
    for (int t = 0; t < NT; ++t) acc[t] = vz;

    const bf16* Arow = A + (size_t)(m0 + r) * K + 8 * hi;
    const bf16* B0   = BT + (size_t)(n0 + r) * K + 8 * hi;
    const size_t bs  = (size_t)16 * K;                   // next N-subtile row

    FragAB aA, aB;
    FragAB bA[NT], bB[NT];

    auto ldA = [&](FragAB& f, int kk) {
        f.h[0] = *(const v8bf*)(Arow + kk);
        f.h[1] = *(const v8bf*)(Arow + kk + 16);
    };
    auto ldB = [&](FragAB (&f)[NT], int kk) {
#pragma unroll
        for (int t = 0; t < NT; ++t) {
            const bf16* p = B0 + (size_t)t * bs + kk;
            f[t].h[0] = *(const v8bf*)p;
            f[t].h[1] = *(const v8bf*)(p + 16);
        }
    };
    auto mma = [&](const FragAB& a, const FragAB (&b)[NT]) {
#pragma unroll
        for (int t = 0; t < NT; ++t)
            acc[t] = __builtin_amdgcn_wmma_f32_16x16x32_bf16(
                false, a.v, false, b[t].v, (short)0, acc[t], false, false);
    };

    // Software pipeline: compute slab k while loading slab k+32.
    ldA(aA, 0); ldB(bA, 0);
    int k = 0;
    for (; k + 64 < K; k += 64) {
        ldA(aB, k + 32); ldB(bB, k + 32);
        mma(aA, bA);
        ldA(aA, k + 64); ldB(bA, k + 64);
        mma(aB, bB);
    }
    // k == K - 64 here (K % 64 == 0)
    ldA(aB, K - 32); ldB(bB, K - 32);
    mma(aA, bA);
    mma(aB, bB);

#pragma unroll
    for (int t = 0; t < NT; ++t) {
        int col = n0 + t * 16 + r;
        float bv = bias ? bias[col] : 0.0f;
#pragma unroll
        for (int v = 0; v < 8; ++v) {
            int row = m0 + v + 8 * hi;
            size_t idx = (size_t)row * N + col;
            float val = acc[t][v] + bv;
            if (Csrc) val += Csrc[idx];
            if (relu) val = fmaxf(val, 0.0f);
            C[idx] = val;
        }
    }
}

// ---------------------------------------------------------------------------
// LDS-tiled transpose: W (K,N) fp32 row-major [optionally + U] -> WT (N,K) bf16.
// 32x32 tile per block; both global reads and writes fully coalesced
// (the naive version read columns at 16KB lane stride -> ~30x wasted HBM BW
// on ~250MB of weights per call).
// ---------------------------------------------------------------------------
__global__ __launch_bounds__(256)
void k_prep_wT(const float* __restrict__ W, const float* __restrict__ U,
               bf16* __restrict__ WT, int K, int N, int rowOff)
{
    __shared__ float tile[32][33];
    int k0 = blockIdx.y * 32;
    int n0 = blockIdx.x * 32;
    int tx = threadIdx.x & 31;
    int ty = threadIdx.x >> 5;          // 0..7
#pragma unroll
    for (int c = 0; c < 4; ++c) {
        int kk = k0 + ty + c * 8;
        float v = W[(size_t)(rowOff + kk) * N + (n0 + tx)];
        if (U) v += U[(size_t)kk * N + (n0 + tx)];
        tile[ty + c * 8][tx] = v;
    }
    __syncthreads();
#pragma unroll
    for (int c = 0; c < 4; ++c) {
        int n = n0 + ty + c * 8;
        WT[(size_t)n * K + (k0 + tx)] = (bf16)tile[tx][ty + c * 8];
    }
}

__global__ void k_f32_to_bf16(const float* __restrict__ in, bf16* __restrict__ out, int n)
{
    int i = blockIdx.x * 256 + threadIdx.x;
    if (i < n) out[i] = (bf16)in[i];
}

// src embeddings, time-major: out[(s*B+b)*E + e] = emb[ids[b*S+s]*E + e]
__global__ void k_gather_src(const int* __restrict__ ids, const float* __restrict__ emb,
                             bf16* __restrict__ out)
{
    int idx = blockIdx.x * 256 + threadIdx.x;     // < S*B*E
    int e = idx % Ee, sb = idx / Ee;
    int s = sb / Bz, b = sb % Bz;
    out[idx] = (bf16)emb[(size_t)ids[b * Ss + s] * Ee + e];
}

// tgt embeddings for steps 0..T-2, time-major rows t*B+b
__global__ void k_gather_tgt(const int* __restrict__ ids, const float* __restrict__ emb,
                             bf16* __restrict__ out)
{
    int idx = blockIdx.x * 256 + threadIdx.x;     // < (T-1)*B*E
    int e = idx % Ee, tb = idx / Ee;
    int t = tb / Bz, b = tb % Bz;
    out[idx] = (bf16)emb[(size_t)ids[b * Tt + t] * Ee + e];
}

__global__ void k_gather_persona(const int* __restrict__ ids, const float* __restrict__ emb,
                                 bf16* __restrict__ out)
{
    int idx = blockIdx.x * 256 + threadIdx.x;     // < B*E
    int e = idx % Ee, b = idx / Ee;
    out[idx] = (bf16)emb[(size_t)ids[b] * Ee + e];
}

// enc (S,B,E) f32 -> enc_flat (B, S*E) bf16  (swapaxes + reshape)
__global__ void k_build_encflat(const float* __restrict__ enc, bf16* __restrict__ out)
{
    int idx = blockIdx.x * 256 + threadIdx.x;     // < B*S*E
    int b = idx / (Ss * Ee);
    int rem = idx % (Ss * Ee);
    int s = rem / Ee, e = rem % Ee;
    out[idx] = (bf16)enc[((size_t)s * Bz + b) * Ee + e];
}

// Keras-style gates on z (B,4H) [i|f|g|o], activation=relu, recurrent=sigmoid.
// c updated in place (f32); h written as bf16 (next GEMM operand).
__global__ void k_lstm_gates(const float* __restrict__ z, const float* __restrict__ zextra,
                             float* __restrict__ c, bf16* __restrict__ h_out)
{
    int idx = blockIdx.x * 256 + threadIdx.x;     // < B*H
    int b = idx >> 10, j = idx & (Hh - 1);
    const float* zr = z + (size_t)b * H4;
    float zi = zr[j], zf = zr[Hh + j], zg = zr[2 * Hh + j], zo = zr[3 * Hh + j];
    if (zextra) {
        const float* er = zextra + (size_t)b * H4;
        zi += er[j]; zf += er[Hh + j]; zg += er[2 * Hh + j]; zo += er[3 * Hh + j];
    }
    float cn = sigmoidf(zf) * c[idx] + sigmoidf(zi) * fmaxf(zg, 0.0f);
    float hn = sigmoidf(zo) * fmaxf(cn, 0.0f);
    c[idx] = cn;
    h_out[idx] = (bf16)hn;
}

// Row softmax (B rows of V) -> probs, plus per-(t,b) xent term (clip like ref).
__global__ void k_softmax_loss(const float* __restrict__ logits, float* __restrict__ probs,
                               const int* __restrict__ tgt, int tstep,
                               float* __restrict__ lossbuf)
{
    __shared__ float red[256];
    int b = blockIdx.x;
    const float* row = logits + (size_t)b * Vv;
    float mx = -1e30f;
    for (int i = threadIdx.x; i < Vv; i += 256) mx = fmaxf(mx, row[i]);
    red[threadIdx.x] = mx; __syncthreads();
    for (int s = 128; s > 0; s >>= 1) {
        if (threadIdx.x < s) red[threadIdx.x] = fmaxf(red[threadIdx.x], red[threadIdx.x + s]);
        __syncthreads();
    }
    mx = red[0]; __syncthreads();
    float sum = 0.0f;
    for (int i = threadIdx.x; i < Vv; i += 256) sum += __expf(row[i] - mx);
    red[threadIdx.x] = sum; __syncthreads();
    for (int s = 128; s > 0; s >>= 1) {
        if (threadIdx.x < s) red[threadIdx.x] += red[threadIdx.x + s];
        __syncthreads();
    }
    sum = red[0];
    float inv = 1.0f / sum;
    float* prow = probs + (size_t)b * Vv;
    for (int i = threadIdx.x; i < Vv; i += 256) prow[i] = __expf(row[i] - mx) * inv;
    if (threadIdx.x == 0) {
        int label = tgt[b * Tt + tstep + 1];
        float p = __expf(row[label] - mx) * inv;
        p = fminf(fmaxf(p, 1e-7f), 1.0f);
        lossbuf[tstep * Bz + b] = -__logf(p);
    }
}

// Deterministic final reduction of (T-1)*B loss terms -> d_out[0].
__global__ void k_loss_reduce(const float* __restrict__ lb, float* __restrict__ out, int n)
{
    __shared__ float red[256];
    float s = 0.0f;
    for (int i = threadIdx.x; i < n; i += 256) s += lb[i];
    red[threadIdx.x] = s; __syncthreads();
    for (int k = 128; k > 0; k >>= 1) {
        if (threadIdx.x < k) red[threadIdx.x] += red[threadIdx.x + k];
        __syncthreads();
    }
    if (threadIdx.x == 0) out[0] = red[0];
}

// ---------------------------------------------------------------------------
extern "C" void kernel_launch(void* const* d_in, const int* in_sizes, int n_in,
                              void* d_out, int out_size, void* d_ws, size_t ws_size,
                              hipStream_t stream)
{
    (void)in_sizes; (void)n_in; (void)out_size; (void)ws_size;

    // ---- inputs per setup_inputs() dict order (params flattened recursively)
    const int*   src     = (const int*)d_in[0];   // (B,S)
    const int*   tgtids  = (const int*)d_in[1];   // (B,T)
    const int*   spkids  = (const int*)d_in[2];   // (B,)
    // d_in[3] addressee_list: unused by reference
    const float* h0      = (const float*)d_in[4];
    const float* c0      = (const float*)d_in[5];
    auto P = [&](int i) { return (const float*)d_in[6 + i]; };
    // enc[l]: W=P(3l) U=P(3l+1) b=P(3l+2);  dec[l]: W=P(12+3l) ...
    const float* d1W = P(24); const float* d1b = P(25);
    const float* d2W = P(26); const float* d2b = P(27);
    const float* doW = P(28); const float* dob = P(29);
    const float* src_emb = P(30); const float* tgt_emb = P(31); const float* persona = P(32);
    float* out = (float*)d_out;                    // [loss | probs (T-1,B,V)]

    // ---- workspace bump allocator (~330 MB total)
    char* wsb = (char*)d_ws; size_t off = 0;
    auto alloc = [&](size_t bytes) -> void* {
        off = (off + 255) & ~(size_t)255;
        void* p = wsb + off; off += bytes; return p;
    };

    bf16* encWT[4]; bf16* encUT[4];
    encWT[0] = (bf16*)alloc((size_t)H4 * Ee * 2);
    for (int l = 1; l < 4; ++l) encWT[l] = (bf16*)alloc((size_t)H4 * Hh * 2);
    for (int l = 0; l < 4; ++l) encUT[l] = (bf16*)alloc((size_t)H4 * Hh * 2);
    bf16* WencT = (bf16*)alloc((size_t)H4 * (Ss * Ee) * 2);   // 84 MB
    bf16* WxT   = (bf16*)alloc((size_t)H4 * Ee * 2);
    bf16* WspkT = (bf16*)alloc((size_t)H4 * Ee * 2);
    bf16* U0T   = (bf16*)alloc((size_t)H4 * Hh * 2);
    bf16* WsumT[3];
    for (int l = 0; l < 3; ++l) WsumT[l] = (bf16*)alloc((size_t)H4 * Hh * 2);
    bf16* d1WT = (bf16*)alloc((size_t)Hh * Hh * 2);
    bf16* d2WT = (bf16*)alloc((size_t)Ee * Hh * 2);
    bf16* doWT = (bf16*)alloc((size_t)Vv * Hh * 2);           // 64 MB

    bf16*  Xsrc   = (bf16*)alloc((size_t)Ss * Bz * Ee * 2);
    bf16*  Ya     = (bf16*)alloc((size_t)Ss * Bz * Hh * 2);
    bf16*  Yb     = (bf16*)alloc((size_t)Ss * Bz * Hh * 2);
    float* ZX     = (float*)alloc((size_t)Ss * Bz * H4 * 4);  // 21 MB
    float* zstep  = (float*)alloc((size_t)Bz * H4 * 4);
    float* cbuf   = (float*)alloc((size_t)Bz * Hh * 4);
    bf16*  h0bf   = (bf16*)alloc((size_t)Bz * Hh * 2);
    float* D1f    = (float*)alloc((size_t)Ss * Bz * Hh * 4);
    bf16*  D1b    = (bf16*)alloc((size_t)Ss * Bz * Hh * 2);
    float* ENCf   = (float*)alloc((size_t)Ss * Bz * Ee * 4);
    bf16*  encfl  = (bf16*)alloc((size_t)Bz * Ss * Ee * 2);
    bf16*  SPK    = (bf16*)alloc((size_t)Bz * Ee * 2);
    bf16*  XT     = (bf16*)alloc((size_t)(Tt - 1) * Bz * Ee * 2);
    float* Zpre   = (float*)alloc((size_t)Bz * H4 * 4);
    float* ZXd    = (float*)alloc((size_t)(Tt - 1) * Bz * H4 * 4); // 20 MB
    bf16*  hD     = (bf16*)alloc((size_t)Bz * Hh * 2);
    float* LOG    = (float*)alloc((size_t)Bz * Vv * 4);
    float* lossbf = (float*)alloc((size_t)(Tt - 1) * Bz * 4);

    // ---- launch helpers
    auto gemm = [&](float* C, const bf16* A, const bf16* Bt, int M, int N, int K,
                    const float* Csrc, const float* bias, int relu) {
        int waves = (M / 16) * (N / 64);
        k_wmma_gemm<<<dim3((waves + 7) / 8), dim3(256), 0, stream>>>(
            A, Bt, C, Csrc, bias, M, N, K, relu);
    };
    auto prep = [&](bf16* WT, const float* W, const float* U, int K, int N, int rowOff) {
        k_prep_wT<<<dim3(N / 32, K / 32), dim3(256), 0, stream>>>(W, U, WT, K, N, rowOff);
    };

    // ---- weight prep (bf16, transposed; re-done every call: deterministic)
    prep(encWT[0], P(0), nullptr, Ee, H4, 0);
    for (int l = 1; l < 4; ++l) prep(encWT[l], P(3 * l), nullptr, Hh, H4, 0);
    for (int l = 0; l < 4; ++l) prep(encUT[l], P(3 * l + 1), nullptr, Hh, H4, 0);
    prep(WencT, P(12), nullptr, Ss * Ee, H4, 0);
    prep(WxT,   P(12), nullptr, Ee, H4, Ss * Ee);
    prep(WspkT, P(12), nullptr, Ee, H4, Ss * Ee + Ee);
    prep(U0T,   P(13), nullptr, Hh, H4, 0);
    for (int l = 1; l < 4; ++l) prep(WsumT[l - 1], P(12 + 3 * l), P(12 + 3 * l + 1), Hh, H4, 0);
    prep(d1WT, d1W, nullptr, Hh, Hh, 0);
    prep(d2WT, d2W, nullptr, Hh, Ee, 0);
    prep(doWT, doW, nullptr, Hh, Vv, 0);

    // ---- gathers + initial state
    k_gather_src<<<dim3(Ss * Bz * Ee / 256), dim3(256), 0, stream>>>(src, src_emb, Xsrc);
    k_gather_tgt<<<dim3((Tt - 1) * Bz * Ee / 256), dim3(256), 0, stream>>>(tgtids, tgt_emb, XT);
    k_gather_persona<<<dim3(Bz * Ee / 256), dim3(256), 0, stream>>>(spkids, persona, SPK);
    k_f32_to_bf16<<<dim3(Bz * Hh / 256), dim3(256), 0, stream>>>(h0, h0bf, Bz * Hh);
    hipMemcpyAsync(cbuf, c0, (size_t)Bz * Hh * 4, hipMemcpyDeviceToDevice, stream);

    // ---- encoder: 4 stacked relu-LSTMs, (h,c) chained across layers
    const bf16* Xin = Xsrc;
    bf16* Y = Ya;
    const bf16* hprev = h0bf;
    int Kin = Ee;
    for (int l = 0; l < 4; ++l) {
        // batched input GEMM over all 20 steps (bias folded here)
        gemm(ZX, Xin, encWT[l], Ss * Bz, H4, Kin, nullptr, P(3 * l + 2), 0);
        for (int t = 0; t < Ss; ++t) {
            gemm(zstep, hprev, encUT[l], Bz, H4, Hh, ZX + (size_t)t * Bz * H4, nullptr, 0);
            k_lstm_gates<<<dim3(Bz * Hh / 256), dim3(256), 0, stream>>>(
                zstep, nullptr, cbuf, Y + (size_t)t * Bz * Hh);
            hprev = Y + (size_t)t * Bz * Hh;
        }
        Xin = Y;
        Y = (Y == Ya) ? Yb : Ya;
        Kin = Hh;
    }

    // ---- dense head: relu(ys@d1W+d1b)@d2W+d2b -> enc_flat
    gemm(D1f, Xin, d1WT, Ss * Bz, Hh, Hh, nullptr, d1b, 1);
    k_f32_to_bf16<<<dim3(Ss * Bz * Hh / 256), dim3(256), 0, stream>>>(D1f, D1b, Ss * Bz * Hh);
    gemm(ENCf, D1b, d2WT, Ss * Bz, Ee, Hh, nullptr, d2b, 0);
    k_build_encflat<<<dim3(Bz * Ss * Ee / 256), dim3(256), 0, stream>>>(ENCf, encfl);

    // ---- decoder constant parts (once, not per step)
    gemm(Zpre, encfl, WencT, Bz, H4, Ss * Ee, nullptr, P(14), 0);       // enc_flat@W + b0
    gemm(Zpre, SPK, WspkT, Bz, H4, Ee, Zpre, nullptr, 0);               // += spk@W
    gemm(ZXd, XT, WxT, (Tt - 1) * Bz, H4, Ee, nullptr, nullptr, 0);     // all x_t@W batched
    hipMemcpyAsync(hD, hprev, (size_t)Bz * Hh * 2, hipMemcpyDeviceToDevice, stream);

    // ---- decoder steps: only h@U (and Wsum for layers 2-4) per step
    for (int t = 0; t < Tt - 1; ++t) {
        gemm(zstep, hD, U0T, Bz, H4, Hh, ZXd + (size_t)t * Bz * H4, nullptr, 0);
        k_lstm_gates<<<dim3(Bz * Hh / 256), dim3(256), 0, stream>>>(zstep, Zpre, cbuf, hD);
        for (int l = 1; l < 4; ++l) {
            gemm(zstep, hD, WsumT[l - 1], Bz, H4, Hh, nullptr, P(12 + 3 * l + 2), 0);
            k_lstm_gates<<<dim3(Bz * Hh / 256), dim3(256), 0, stream>>>(zstep, nullptr, cbuf, hD);
        }
        gemm(LOG, hD, doWT, Bz, Vv, Hh, nullptr, dob, 0);
        k_softmax_loss<<<dim3(Bz), dim3(256), 0, stream>>>(
            LOG, out + 1 + (size_t)t * Bz * Vv, tgtids, t, lossbf);
    }
    k_loss_reduce<<<dim3(1), dim3(256), 0, stream>>>(lossbf, out, (Tt - 1) * Bz);
}
